// GATActor_21612275434425
// MI455X (gfx1250) — matrix operand order repaired and checked by reading, not compiled, not run on hardware
//
#include <hip/hip_runtime.h>
#include <hip/hip_bf16.h>

#define HEADS   4
#define HID     256
#define HH      1024      // HEADS*HID
#define IN_DIM  131
#define K0PAD   160       // IN_DIM padded to multiple of 32
#define OUT_DIM 6
#define NEG_SLOPE 0.2f

typedef __bf16 bf16_t;
typedef bf16_t bf16x8  __attribute__((ext_vector_type(8)));
typedef bf16_t bf16x16 __attribute__((ext_vector_type(16)));
typedef float  f32x8   __attribute__((ext_vector_type(8)));

// ---------- helpers ----------
__device__ __forceinline__ bf16_t f2bf(float f) {
    union { float f; unsigned u; } in; in.f = f;
    unsigned u = in.u;
    u += 0x7FFFu + ((u >> 16) & 1u);          // round-to-nearest-even
    union { unsigned short s; bf16_t b; } out;
    out.s = (unsigned short)(u >> 16);
    return out.b;
}

// monotonic uint encoding of float for atomicMax (all encodings > 0, so 0 == -inf sentinel)
__device__ __forceinline__ unsigned ord_encode(float f) {
    unsigned u = __float_as_uint(f);
    return (u & 0x80000000u) ? ~u : (u | 0x80000000u);
}
__device__ __forceinline__ float ord_decode(unsigned u) {
    return __uint_as_float((u & 0x80000000u) ? (u & 0x7FFFFFFFu) : ~u);
}
__device__ __forceinline__ float lrelu(float v) { return v > 0.0f ? v : NEG_SLOPE * v; }

// ---------- zero fill ----------
__global__ void zero_kernel(float* __restrict__ p, int n) {
    int i = blockIdx.x * blockDim.x + threadIdx.x;
    if (i < n) p[i] = 0.0f;
}

// ---------- pack input x [N,131] f32 -> [N,160] bf16 (zero padded) ----------
__global__ void pack_x0_kernel(const float* __restrict__ x, bf16_t* __restrict__ xbf, int N) {
    int idx = blockIdx.x * blockDim.x + threadIdx.x;
    if (idx >= N * K0PAD) return;
    int n = idx / K0PAD, c = idx % K0PAD;
    float v = (c < IN_DIM) ? x[(size_t)n * IN_DIM + c] : 0.0f;
    xbf[idx] = f2bf(v);
}

// ---------- W [K,1024] f32 -> Wt [1024,Kpad] bf16 (transposed, zero padded) ----------
__global__ void transpose_w_kernel(const float* __restrict__ W, bf16_t* __restrict__ Wt,
                                   int K, int Kpad) {
    int idx = blockIdx.x * blockDim.x + threadIdx.x;
    if (idx >= HH * Kpad) return;
    int n = idx / Kpad, k = idx % Kpad;
    float v = (k < K) ? W[(size_t)k * HH + n] : 0.0f;
    Wt[idx] = f2bf(v);
}

// ---------- GEMM: C[N,1024] = A[N,K](bf16) x Bt[1024,K](bf16, B transposed), f32 accum ----------
__global__ __launch_bounds__(256)
void gemm_wmma_bf16_kernel(const bf16_t* __restrict__ A, const bf16_t* __restrict__ Bt,
                           float* __restrict__ C, int Nrows, int K) {
    const int lane    = threadIdx.x & 31;
    const int wave    = threadIdx.x >> 5;     // 0..7
    const int l16     = lane & 15;
    const int halfsel = lane >> 4;            // 0 | 1
    const int bm = blockIdx.x * 128;
    const int bn = blockIdx.y * 128;

    int arow = bm + wave * 16 + l16;
    int arow_c = arow < Nrows ? arow : Nrows - 1;      // clamp loads; stores are guarded
    const bf16_t* arowp = A + (size_t)arow_c * K;

    f32x8 acc[8];
    #pragma unroll
    for (int t = 0; t < 8; ++t) acc[t] = (f32x8){0,0,0,0,0,0,0,0};

    for (int k0 = 0; k0 < K; k0 += 32) {
        // A fragment: 16x32 bf16 (lanes<16: K [k0,k0+8)+[k0+16,k0+24); lanes>=16: +8)
        const int ka = k0 + halfsel * 8;
        bf16x8 alo = *(const bf16x8*)(arowp + ka);
        bf16x8 ahi = *(const bf16x8*)(arowp + ka + 16);
        bf16x16 afrag = __builtin_shufflevector(alo, ahi,
            0,1,2,3,4,5,6,7,8,9,10,11,12,13,14,15);

        const int kb = k0 + halfsel * 8;      // B: same K order as A
        #pragma unroll
        for (int t = 0; t < 8; ++t) {
            int col = bn + t * 16 + l16;
            bf16x8 blo = *(const bf16x8*)(Bt + (size_t)col * K + kb);
            bf16x8 bhi = *(const bf16x8*)(Bt + (size_t)col * K + kb + 16);
            bf16x16 bfrag = __builtin_shufflevector(blo, bhi,
                0,1,2,3,4,5,6,7,8,9,10,11,12,13,14,15);
            acc[t] = __builtin_amdgcn_wmma_f32_16x16x32_bf16(
                false, afrag, false, bfrag, (short)0, acc[t], false, false);
        }
    }

    // C layout: VGPR r -> row r + 8*halfsel, col = lane&15
    const int rowbase = bm + wave * 16 + halfsel * 8;
    #pragma unroll
    for (int t = 0; t < 8; ++t) {
        int col = bn + t * 16 + l16;
        #pragma unroll
        for (int r = 0; r < 8; ++r) {
            int row = rowbase + r;
            if (row < Nrows) C[(size_t)row * HH + col] = acc[t][r];
        }
    }
}

// ---------- attention scores: one wave per (node, head) ----------
__global__ __launch_bounds__(256)
void scores_kernel(const float* __restrict__ h, const float* __restrict__ a_src,
                   const float* __restrict__ a_dst, float* __restrict__ s_src,
                   float* __restrict__ s_dst, int N) {
    int wid  = (blockIdx.x * blockDim.x + threadIdx.x) >> 5;
    int lane = threadIdx.x & 31;
    if (wid >= N * HEADS) return;
    int n = wid >> 2, hd = wid & 3;
    const float* hp = h + (size_t)n * HH + hd * HID;
    const float* as = a_src + hd * HID;
    const float* ad = a_dst + hd * HID;
    float ss = 0.0f, sd = 0.0f;
    #pragma unroll
    for (int i = 0; i < HID / 32; ++i) {
        int d = lane + i * 32;
        float v = hp[d];
        ss += v * as[d];
        sd += v * ad[d];
    }
    #pragma unroll
    for (int m = 16; m > 0; m >>= 1) {
        ss += __shfl_xor(ss, m, 32);
        sd += __shfl_xor(sd, m, 32);
    }
    if (lane == 0) { s_src[wid] = ss; s_dst[wid] = sd; }
}

// ---------- edge pass 1: segment max (encoded-uint atomicMax) ----------
__global__ void edge_max_kernel(const int* __restrict__ ei, const float* __restrict__ s_src,
                                const float* __restrict__ s_dst, unsigned* __restrict__ m,
                                int E, int N) {
    int idx = blockIdx.x * blockDim.x + threadIdx.x;
    if (idx >= (E + N) * HEADS) return;
    int e = idx >> 2, hd = idx & 3;
    int s, d;
    if (e < E) { s = ei[e]; d = ei[E + e]; } else { s = d = e - E; }
    float v = lrelu(s_src[s * HEADS + hd] + s_dst[d * HEADS + hd]);
    atomicMax(&m[d * HEADS + hd], ord_encode(v));
}

// ---------- edge pass 2: exp + segment sum ----------
__global__ void edge_exp_kernel(const int* __restrict__ ei, const float* __restrict__ s_src,
                                const float* __restrict__ s_dst, const unsigned* __restrict__ m,
                                float* __restrict__ denom, float* __restrict__ exbuf,
                                int E, int N) {
    int idx = blockIdx.x * blockDim.x + threadIdx.x;
    if (idx >= (E + N) * HEADS) return;
    int e = idx >> 2, hd = idx & 3;
    int s, d;
    if (e < E) { s = ei[e]; d = ei[E + e]; } else { s = d = e - E; }
    float v  = lrelu(s_src[s * HEADS + hd] + s_dst[d * HEADS + hd]);
    float ex = __expf(v - ord_decode(m[d * HEADS + hd]));
    exbuf[idx] = ex;
    atomicAdd(&denom[d * HEADS + hd], ex);
}

// ---------- aggregation: one wave per edge, agg[dst] += alpha * h[src] ----------
__global__ __launch_bounds__(256)
void aggregate_kernel(const int* __restrict__ ei, const float* __restrict__ h,
                      const float* __restrict__ exbuf, const float* __restrict__ denom,
                      float* __restrict__ agg, int E, int N) {
    int wid  = (blockIdx.x * blockDim.x + threadIdx.x) >> 5;
    int lane = threadIdx.x & 31;
    if (wid >= E + N) return;
    int s, d;
    if (wid < E) { s = ei[wid]; d = ei[E + wid]; } else { s = d = wid - E; }
    float aval = 0.0f;
    if (lane < HEADS) aval = exbuf[wid * HEADS + lane] / denom[d * HEADS + lane];
    const float* hp = h + (size_t)s * HH;
    float* op = agg + (size_t)d * HH;
    #pragma unroll
    for (int it = 0; it < HH / 128; ++it) {
        int base = it * 128 + lane * 4;
        float alpha = __shfl(aval, base >> 8, 32);   // head = base/HID
        float4 v = *(const float4*)(hp + base);
        atomicAdd(op + base + 0, alpha * v.x);
        atomicAdd(op + base + 1, alpha * v.y);
        atomicAdd(op + base + 2, alpha * v.z);
        atomicAdd(op + base + 3, alpha * v.w);
    }
}

// ---------- post: bias (+ residual ReLU), write f32 in place and bf16 for next GEMM ----------
__global__ void post_kernel(float* __restrict__ agg_x, const float* __restrict__ b,
                            const float* __restrict__ res, bf16_t* __restrict__ xbf,
                            int N, int do_relu) {
    int idx = blockIdx.x * blockDim.x + threadIdx.x;
    if (idx >= N * HH) return;
    int c = idx & (HH - 1);
    float v = agg_x[idx] + b[c];
    if (do_relu) { v += res[idx]; v = v > 0.0f ? v : 0.0f; }
    agg_x[idx] = v;
    xbf[idx] = f2bf(v);
}

// ---------- FC [1024->6] + softmax: one wave per node; out = softmax || logits ----------
__global__ __launch_bounds__(256)
void fc_softmax_kernel(const float* __restrict__ x, const float* __restrict__ Wfc,
                       const float* __restrict__ bfc, float* __restrict__ out, int N) {
    int wid  = (blockIdx.x * blockDim.x + threadIdx.x) >> 5;
    int lane = threadIdx.x & 31;
    if (wid >= N) return;
    const float* xp = x + (size_t)wid * HH;
    float acc[OUT_DIM] = {0,0,0,0,0,0};
    for (int k = lane; k < HH; k += 32) {
        float xv = xp[k];
        const float* wp = Wfc + (size_t)k * OUT_DIM;
        #pragma unroll
        for (int j = 0; j < OUT_DIM; ++j) acc[j] += xv * wp[j];
    }
    #pragma unroll
    for (int j = 0; j < OUT_DIM; ++j)
        #pragma unroll
        for (int m = 16; m > 0; m >>= 1) acc[j] += __shfl_xor(acc[j], m, 32);
    if (lane == 0) {
        float mx = -3.4e38f;
        #pragma unroll
        for (int j = 0; j < OUT_DIM; ++j) { acc[j] += bfc[j]; mx = fmaxf(mx, acc[j]); }
        float se = 0.0f, ex[OUT_DIM];
        #pragma unroll
        for (int j = 0; j < OUT_DIM; ++j) { ex[j] = __expf(acc[j] - mx); se += ex[j]; }
        float inv = 1.0f / se;
        #pragma unroll
        for (int j = 0; j < OUT_DIM; ++j) {
            out[(size_t)wid * OUT_DIM + j] = ex[j] * inv;                       // softmax
            out[(size_t)N * OUT_DIM + (size_t)wid * OUT_DIM + j] = acc[j];      // logits
        }
    }
}

// ---------------------------------------------------------------------------
extern "C" void kernel_launch(void* const* d_in, const int* in_sizes, int n_in,
                              void* d_out, int out_size, void* d_ws, size_t ws_size,
                              hipStream_t stream) {
    const int N = in_sizes[0] / IN_DIM;        // 20000
    const int E = in_sizes[1] / 2;             // 320000
    const int ET = E + N;                      // edges incl. self loops

    const float* x_in = (const float*)d_in[0];
    const int*   ei   = (const int*)d_in[1];
    const float* Ws[3]   = {(const float*)d_in[2],  (const float*)d_in[6],  (const float*)d_in[10]};
    const float* asrc[3] = {(const float*)d_in[3],  (const float*)d_in[7],  (const float*)d_in[11]};
    const float* adst[3] = {(const float*)d_in[4],  (const float*)d_in[8],  (const float*)d_in[12]};
    const float* bs[3]   = {(const float*)d_in[5],  (const float*)d_in[9],  (const float*)d_in[13]};
    const float* Wfc = (const float*)d_in[14];
    const float* bfc = (const float*)d_in[15];
    float* out = (float*)d_out;

    // ---- workspace layout (256B aligned) ----
    char* w = (char*)d_ws;
    size_t off = 0;
    auto alloc = [&](size_t bytes) { void* p = w + off; off = (off + bytes + 255) & ~(size_t)255; return p; };
    const size_t NB = (size_t)N * HH * sizeof(float);
    float*   bufA  = (float*)  alloc(NB);                          // ping
    float*   bufB  = (float*)  alloc(NB);                          // pong
    float*   hbuf  = (float*)  alloc(NB);                          // h = x@W
    bf16_t*  xbf   = (bf16_t*) alloc((size_t)N * HH * sizeof(bf16_t));
    bf16_t*  Wt    = (bf16_t*) alloc((size_t)HH * HH * sizeof(bf16_t));
    float*   ssrc  = (float*)  alloc((size_t)N * HEADS * sizeof(float));
    float*   sdst  = (float*)  alloc((size_t)N * HEADS * sizeof(float));
    unsigned* mbuf = (unsigned*)alloc((size_t)N * HEADS * sizeof(unsigned));
    float*   denom = (float*)  alloc((size_t)N * HEADS * sizeof(float));   // contiguous after mbuf
    float*   exbuf = (float*)  alloc((size_t)ET * HEADS * sizeof(float));
    (void)ws_size; (void)n_in; (void)out_size;

    const int T = 256;
    auto blk = [](long n, int t) { return (unsigned)((n + t - 1) / t); };

    // pack x -> bf16 [N, K0PAD]
    pack_x0_kernel<<<blk((long)N * K0PAD, T), T, 0, stream>>>(x_in, xbf, N);

    float* aggs[3] = {bufA, bufB, bufA};
    float* ress[3] = {nullptr, bufA, bufB};

    for (int i = 0; i < 3; ++i) {
        const int K    = (i == 0) ? K0PAD  : HH;
        const int Kraw = (i == 0) ? IN_DIM : HH;
        float* aggp = aggs[i];

        // W^T -> bf16 [1024, K]
        transpose_w_kernel<<<blk((long)HH * K, T), T, 0, stream>>>(Ws[i], Wt, Kraw, K);

        // h = x @ W   (WMMA bf16, f32 accumulate)
        dim3 grid(blk(N, 128), HH / 128);
        gemm_wmma_bf16_kernel<<<grid, T, 0, stream>>>(xbf, Wt, hbuf, N, K);

        // per-(node,head) attention score dots
        scores_kernel<<<blk((long)N * HEADS * 32, T), T, 0, stream>>>(hbuf, asrc[i], adst[i], ssrc, sdst, N);

        // clear accumulators (mbuf & denom are contiguous -> one clear)
        zero_kernel<<<blk((long)N * HH, T), T, 0, stream>>>(aggp, N * HH);
        zero_kernel<<<blk((long)2 * N * HEADS, T), T, 0, stream>>>((float*)mbuf, 2 * N * HEADS);

        // segment softmax over incoming edges of each dst
        edge_max_kernel<<<blk((long)ET * HEADS, T), T, 0, stream>>>(ei, ssrc, sdst, mbuf, E, N);
        edge_exp_kernel<<<blk((long)ET * HEADS, T), T, 0, stream>>>(ei, ssrc, sdst, mbuf, denom, exbuf, E, N);

        // agg[dst] += alpha * h[src]
        aggregate_kernel<<<blk((long)ET * 32, T), T, 0, stream>>>(ei, hbuf, exbuf, denom, aggp, E, N);

        // bias (+ residual relu for i>0); refresh bf16 input for next GEMM
        post_kernel<<<blk((long)N * HH, T), T, 0, stream>>>(aggp, bs[i], ress[i], xbf, N, i > 0 ? 1 : 0);
    }

    // logits + softmax; final node features live in bufA (aggs[2])
    fc_softmax_kernel<<<blk((long)N * 32, T), T, 0, stream>>>(bufA, Wfc, bfc, out, N);
}
